// DiffusionEnsembleHead_68324339745109
// MI455X (gfx1250) — matrix operand order repaired and checked by reading, not compiled, not run on hardware
//
#include <hip/hip_runtime.h>
#include <hip/hip_bf16.h>

#define N_STEPS 50
#define BETA_START 1e-4f
#define BETA_END 0.02f
#define IN_DIM 512
#define OUT_DIM 128
#define HID 128
#define H2 256
#define SAMPLES 8
#define BATCH 512
#define MTOTAL (SAMPLES * BATCH)   // 4096 rows

typedef _Float16 h16;
typedef __attribute__((ext_vector_type(16))) _Float16 v16h;
typedef __attribute__((ext_vector_type(8)))  _Float16 v8h;
typedef __attribute__((ext_vector_type(8)))  float    v8f;
typedef __attribute__((ext_vector_type(4)))  unsigned int uint4v;

// ---------------- helpers ----------------
// branch-free tanh-form GELU: one v_exp_f32 + one v_rcp_f32
static __device__ __forceinline__ float geluf(float x) {
    float u = 1.5957691216f * (x + 0.044715f * x * x * x);
    float e = __expf(u);                                     // exp(2*z)
    float t = 1.0f - 2.0f * __builtin_amdgcn_rcpf(e + 1.0f); // tanh(z), inf-safe
    return 0.5f * x * (1.0f + t);
}

// exact-erf GELU for the (cold) precompute kernels
static __device__ __forceinline__ float geluf_exact(float x) {
    return 0.5f * x * (1.0f + erff(x * 0.70710678118654752f));
}

static __device__ __forceinline__ unsigned hash32(unsigned s) {
    s ^= s >> 16; s *= 0x7feb352du;
    s ^= s >> 15; s *= 0x846ca68bu;
    s ^= s >> 16; return s;
}

// one Box-Muller -> TWO gaussians (branch-free; v_log/v_sqrt/v_sin/v_cos)
static __device__ __forceinline__ void gauss2f(unsigned a, unsigned b,
                                               float* g0, float* g1) {
    unsigned u = hash32(a ^ 0x68bc21ebu) + b * 0x9E3779B9u;
    unsigned v = hash32(u ^ 0x02e5be93u);
    unsigned w = hash32(v + 0x9E3779B9u);
    float u1 = (float)(v >> 8) * 0x1p-24f + 0x1p-25f;  // (0,1]
    float u2 = (float)(w >> 8) * 0x1p-24f;
    float r  = __builtin_amdgcn_sqrtf(-2.0f * __logf(u1));
    float s, c;
    __sincosf(6.28318530717958f * u2, &s, &c);
    *g0 = r * c;
    *g1 = r * s;
}

static __device__ __forceinline__ float gaussf(unsigned a, unsigned b) {
    float g0, g1;
    gauss2f(a, b, &g0, &g1);
    return g0;
}

static __device__ __forceinline__ v8f wmma16(v16h a, v16h b, v8f c) {
    return __builtin_amdgcn_wmma_f32_16x16x32_f16(false, a, false, b,
                                                  (short)0, c, false, false);
}

// ---------------- workspace layout (bytes) ----------------
#define OFF_STEPVEC  0          // 50*256 f32        = 51200
#define OFF_COEFS    52224      // 150 f32 (rsa|coef|sqb)
#define OFF_W1XT     53248      // 256x128 f16       = 65536
#define OFF_W2T      118784     // 256x256 f16       = 131072
#define OFF_W3T      249856     // 128x256 f16       = 65536
#define OFF_CONDC    315392     // 32*16*32*8 f32    = 524288 (fragment order)
#define OFF_MEAN     839680     // 512*128 f32       = 262144

// ============ precompute A: per-step time-embedding vector + coefs ============
__global__ void pre_time(const float* __restrict__ te_w1, const float* __restrict__ te_b1,
                         const float* __restrict__ te_w2, const float* __restrict__ te_b2,
                         const float* __restrict__ dn_w1, const float* __restrict__ dn_b1,
                         float* __restrict__ stepvec, float* __restrict__ coefs) {
    __shared__ float pre[HID];
    __shared__ float temb[HID];
    const int tid = threadIdx.x;   // 256 threads

    if (tid < N_STEPS) {
        const float db = (BETA_END - BETA_START) / (float)(N_STEPS - 1);
        float beta = BETA_START + db * (float)tid;
        float ac = 1.0f;
        for (int j = 0; j <= tid; ++j) ac *= 1.0f - (BETA_START + db * (float)j);
        coefs[tid]       = rsqrtf(1.0f - beta);          // 1/sqrt(alpha_i)
        coefs[50 + tid]  = beta / sqrtf(1.0f - ac);      // beta / sqrt_omac_i
        coefs[100 + tid] = (tid > 0) ? sqrtf(beta) : 0.0f;
    }

    for (int i = 0; i < N_STEPS; ++i) {
        float t = (float)i / (float)N_STEPS;
        if (tid < HID) pre[tid] = geluf_exact(t * te_w1[tid] + te_b1[tid]);
        __syncthreads();
        if (tid < HID) {
            float s = te_b2[tid];
            for (int h = 0; h < HID; ++h) s += pre[h] * te_w2[h * HID + tid];
            temb[tid] = s;
        }
        __syncthreads();
        {   // stepvec[i][n] = dn_b1[n] + t_emb @ dn_w1[640:768, n]
            float s = dn_b1[tid];
            for (int j = 0; j < HID; ++j) s += temb[j] * dn_w1[(640 + j) * H2 + tid];
            stepvec[i * H2 + tid] = s;
        }
        __syncthreads();
    }
}

// ============ precompute B: transpose + f16-convert denoise weights ============
__global__ void pre_weights(const float* __restrict__ dn_w1, const float* __restrict__ dn_w2,
                            const float* __restrict__ dn_w3,
                            h16* __restrict__ w1xT, h16* __restrict__ w2T, h16* __restrict__ w3T) {
    int idx = blockIdx.x * blockDim.x + threadIdx.x;
    if (idx < 256 * 128) {                       // W1x^T[n][k] = dn_w1[k][n], k<128
        int n = idx >> 7, k = idx & 127;
        w1xT[idx] = (h16)dn_w1[k * H2 + n];
    } else if (idx < 256 * 128 + 256 * 256) {    // W2^T[n][k] = dn_w2[k][n]
        int j = idx - 256 * 128;
        int n = j >> 8, k = j & 255;
        w2T[j] = (h16)dn_w2[k * H2 + n];
    } else if (idx < 256 * 128 + 256 * 256 + 128 * 256) {  // W3^T[n][k] = dn_w3[k][n]
        int j = idx - (256 * 128 + 256 * 256);
        int n = j >> 8, k = j & 255;
        w3T[j] = (h16)dn_w3[k * OUT_DIM + n];
    }
}

// ============ precompute C: cond contribution (C-fragment order) + mean head ============
__global__ void pre_cond(const float* __restrict__ x, const float* __restrict__ dn_w1,
                         const float* __restrict__ mh_w, const float* __restrict__ mh_b,
                         float* __restrict__ condc, float* __restrict__ mean) {
    int idx = blockIdx.x * blockDim.x + threadIdx.x;
    if (idx < 32 * 16 * 32 * 8) {
        // condc[bg][nt][lane][r] = (x @ dn_w1[128:640])[b][n] in WMMA C layout
        int r = idx & 7, lane = (idx >> 3) & 31, nt = (idx >> 8) & 15, bg = idx >> 12;
        int b = bg * 16 + r + ((lane >> 4) << 3);
        int n = nt * 16 + (lane & 15);
        float s = 0.0f;
        for (int k = 0; k < IN_DIM; ++k) s += x[b * IN_DIM + k] * dn_w1[(128 + k) * H2 + n];
        condc[idx] = s;
    } else if (idx < 131072 + BATCH * OUT_DIM) {
        int j = idx - 131072;
        int b = j >> 7, o = j & 127;
        float s = mh_b[o];
        for (int k = 0; k < IN_DIM; ++k) s += x[b * IN_DIM + k] * mh_w[k * OUT_DIM + o];
        mean[j] = s;
    }
}

// ============ main: persistent diffusion loop, all WMMA from LDS ============
// grid = 64 WGs x 256 threads (8 waves -> 2 waves/SIMD for latency hiding).
// Wave PAIRS share a 16-row band and split the N tiles; 64 rows per WG.
// LDS: W1x^T(64K) + W2^T(128K) + W3^T(64K) + x_t f16(16K) + h f16(32K) = 304 KB/WGP.
// Step-invariant operands (cond fragments, biases) live in VGPRs for all 50 steps.
// B fragments are software-pipelined one WMMA ahead to hide LDS latency.
__global__ __launch_bounds__(256) void diff_main(
    const float* __restrict__ stepvec, const float* __restrict__ coefs,
    const h16* __restrict__ w1xT, const h16* __restrict__ w2T, const h16* __restrict__ w3T,
    const float* __restrict__ condc, const float* __restrict__ mean,
    const float* __restrict__ dn_b2, const float* __restrict__ dn_b3,
    float* __restrict__ out) {
    extern __shared__ char smem[];
    h16* sW1 = (h16*)smem;             // [256][128]
    h16* sW2 = sW1 + 256 * 128;        // [256][256]
    h16* sW3 = sW2 + 256 * 256;        // [128][256]
    h16* sX  = sW3 + 128 * 256;        // [64][128]  x_t
    h16* sH  = sX + 64 * 128;          // [64][256]  hidden

    const int tid = threadIdx.x;
    {   // stage all weights to LDS once (b128 copies)
        const uint4v* g1 = (const uint4v*)w1xT; uint4v* l1 = (uint4v*)sW1;
        for (int i = tid; i < 256 * 128 / 8; i += 256) l1[i] = g1[i];
        const uint4v* g2 = (const uint4v*)w2T;  uint4v* l2 = (uint4v*)sW2;
        for (int i = tid; i < 256 * 256 / 8; i += 256) l2[i] = g2[i];
        const uint4v* g3 = (const uint4v*)w3T;  uint4v* l3 = (uint4v*)sW3;
        for (int i = tid; i < 128 * 256 / 8; i += 256) l3[i] = g3[i];
    }

    const int lane = tid & 31;
    const int wv   = tid >> 5;                // 0..7
    const int pair = wv >> 1;                 // 0..3 : 16-row band
    const int half = wv & 1;                  // N-tile split within the pair
    const int m0   = pair << 4;               // band base row in WG tile
    const int row0 = blockIdx.x * 64;         // WG's global row base
    const int mrow = lane & 15;               // A-row / C-col index
    const int hi   = lane >> 4;               // half-wave select
    const int kb   = hi << 4;                 // A fragment K base (0/16)
    const int bg   = ((row0 >> 4) + pair) & 31; // batch row-group (repeats every 512 rows)

    // ---- hoist step-invariant operands into registers ----
    v8f ccond[8];                              // this wave's 8 GEMM1 tiles (C layout)
#pragma unroll
    for (int nt = 0; nt < 8; ++nt) {
        const int ntg = (half << 3) + nt;
        ccond[nt] = *(const v8f*)&condc[(((bg << 4) + ntg) * 32 + lane) * 8];
    }
    float bb2[8], bb3[4];
#pragma unroll
    for (int nt = 0; nt < 8; ++nt) bb2[nt] = dn_b2[((half << 3) + nt) * 16 + mrow];
#pragma unroll
    for (int nt = 0; nt < 4; ++nt) bb3[nt] = dn_b3[((half << 2) + nt) * 16 + mrow];

    // init x_T ~ N(0,1); wave pair splits the band's 2048 elements
    for (int e = (half << 5) + lane; e < 16 * 128; e += 64) {
        int r = e >> 7, c = e & 127;
        int grow = row0 + m0 + r;
        sX[(m0 + r) * 128 + c] = (h16)gaussf(0xA0A0A0A0u ^ (unsigned)grow, (unsigned)c + 0x1234567u);
    }
    __syncthreads();   // weights staged + x_t initialized

    for (int i = N_STEPS - 1; i >= 0; --i) {
        // per-step scalars / vectors (issued together)
        const float rsa = coefs[i];
        const float cf  = coefs[50 + i];
        const float sqb = coefs[100 + i];   // 0 for i==0
        const float* svp = &stepvec[i * H2];
        float sv[8];
#pragma unroll
        for (int nt = 0; nt < 8; ++nt) sv[nt] = svp[((half << 3) + nt) * 16 + mrow];

        // ---------- GEMM1: x_t(16x128) @ W1x -> h cols [half*128, half*128+128) ----------
        v16h a1[4];
#pragma unroll
        for (int kk = 0; kk < 4; ++kk) {
            const h16* ap = &sX[(m0 + mrow) * 128 + kk * 32 + kb];
            v8h lo = *(const v8h*)ap;
            v8h hv = *(const v8h*)(ap + 8);
            a1[kk] = __builtin_shufflevector(lo, hv, 0,1,2,3,4,5,6,7,8,9,10,11,12,13,14,15);
        }
#pragma unroll
        for (int nt = 0; nt < 8; ++nt) {
            const int ncol = ((half << 3) + nt) * 16 + mrow;
            const h16* bbase = &sW1[ncol * 128 + (hi << 4)];
            v16h bcur = *(const v16h*)bbase;
            v8f c = ccond[nt];
#pragma unroll
            for (int r = 0; r < 8; ++r) c[r] += sv[nt];
#pragma unroll
            for (int kk = 0; kk < 4; ++kk) {          // B pipelined one WMMA ahead
                v16h bnxt = bcur;
                if (kk < 3) bnxt = *(const v16h*)(bbase + (kk + 1) * 32);
                c = wmma16(a1[kk], bcur, c);
                bcur = bnxt;
            }
#pragma unroll
            for (int r = 0; r < 8; ++r)
                sH[(m0 + r + (hi << 3)) * H2 + ncol] = (h16)geluf(c[r]);
        }
        __syncthreads();   // partner's h columns complete

        // ---------- GEMM2: h(16x256) @ W2 -> h' cols [half*128, ...) ----------
        v16h a2[8];
#pragma unroll
        for (int kk = 0; kk < 8; ++kk) {
            const h16* ap = &sH[(m0 + mrow) * H2 + kk * 32 + kb];
            v8h lo = *(const v8h*)ap;
            v8h hv = *(const v8h*)(ap + 8);
            a2[kk] = __builtin_shufflevector(lo, hv, 0,1,2,3,4,5,6,7,8,9,10,11,12,13,14,15);
        }
        __syncthreads();   // everyone's A fragments in registers -> safe to overwrite sH
#pragma unroll
        for (int nt = 0; nt < 8; ++nt) {
            const int ncol = ((half << 3) + nt) * 16 + mrow;
            const h16* bbase = &sW2[ncol * H2 + (hi << 4)];
            v16h bcur = *(const v16h*)bbase;
            v8f c;
#pragma unroll
            for (int r = 0; r < 8; ++r) c[r] = bb2[nt];
#pragma unroll
            for (int kk = 0; kk < 8; ++kk) {
                v16h bnxt = bcur;
                if (kk < 7) bnxt = *(const v16h*)(bbase + (kk + 1) * 32);
                c = wmma16(a2[kk], bcur, c);
                bcur = bnxt;
            }
#pragma unroll
            for (int r = 0; r < 8; ++r)
                sH[(m0 + r + (hi << 3)) * H2 + ncol] = (h16)geluf(c[r]);
        }
        __syncthreads();   // partner's h' columns complete

        // ---------- GEMM3: h'(16x256) @ W3 -> noise_pred cols, fused x_t update ----------
        v16h a3[8];
#pragma unroll
        for (int kk = 0; kk < 8; ++kk) {
            const h16* ap = &sH[(m0 + mrow) * H2 + kk * 32 + kb];
            v8h lo = *(const v8h*)ap;
            v8h hv = *(const v8h*)(ap + 8);
            a3[kk] = __builtin_shufflevector(lo, hv, 0,1,2,3,4,5,6,7,8,9,10,11,12,13,14,15);
        }
        __syncthreads();   // sH free; also x_t reads below precede partner's writes
#pragma unroll
        for (int nt = 0; nt < 4; ++nt) {
            const int ncol = ((half << 2) + nt) * 16 + mrow;
            const h16* bbase = &sW3[ncol * H2 + (hi << 4)];
            v16h bcur = *(const v16h*)bbase;
            v8f c;
#pragma unroll
            for (int r = 0; r < 8; ++r) c[r] = bb3[nt];
#pragma unroll
            for (int kk = 0; kk < 8; ++kk) {
                v16h bnxt = bcur;
                if (kk < 7) bnxt = *(const v16h*)(bbase + (kk + 1) * 32);
                c = wmma16(a3[kk], bcur, c);
                bcur = bnxt;
            }
            // x_t update: one Box-Muller per row pair (2 gaussians per transcendental set)
#pragma unroll
            for (int rp = 0; rp < 4; ++rp) {
                const int r0   = rp << 1;
                const int lr0  = m0 + r0 + (hi << 3);
                const int grow = row0 + lr0;
                float g0, g1;
                gauss2f((unsigned)i * 0x01000193u ^ (unsigned)grow * 0x9E3779B9u,
                        (unsigned)ncol * 0x85EBCA6Bu + 0xC2B2AE35u, &g0, &g1);
                float xt0 = (float)sX[lr0 * 128 + ncol];
                float xt1 = (float)sX[(lr0 + 1) * 128 + ncol];
                sX[lr0 * 128 + ncol]       = (h16)(rsa * (xt0 - cf * c[r0])     + sqb * g0);
                sX[(lr0 + 1) * 128 + ncol] = (h16)(rsa * (xt1 - cf * c[r0 + 1]) + sqb * g1);
            }
        }
        __syncthreads();   // partner's x_t columns complete before next step's A loads
    }

    // ---------- final output pass: out = x_0 + mean (coalesced) ----------
    for (int e = (half << 5) + lane; e < 16 * 128; e += 64) {
        int r = e >> 7, c = e & 127;
        int grow = row0 + m0 + r;
        out[grow * OUT_DIM + c] =
            (float)sX[(m0 + r) * 128 + c] + mean[(grow & 511) * OUT_DIM + c];
    }
}

// ---------------- host launcher ----------------
extern "C" void kernel_launch(void* const* d_in, const int* in_sizes, int n_in,
                              void* d_out, int out_size, void* d_ws, size_t ws_size,
                              hipStream_t stream) {
    const float* x     = (const float*)d_in[0];
    // d_in[1] = n_samples (==8, hardcoded)
    const float* te_w1 = (const float*)d_in[2];
    const float* te_b1 = (const float*)d_in[3];
    const float* te_w2 = (const float*)d_in[4];
    const float* te_b2 = (const float*)d_in[5];
    const float* dn_w1 = (const float*)d_in[6];
    const float* dn_b1 = (const float*)d_in[7];
    const float* dn_w2 = (const float*)d_in[8];
    const float* dn_b2 = (const float*)d_in[9];
    const float* dn_w3 = (const float*)d_in[10];
    const float* dn_b3 = (const float*)d_in[11];
    const float* mh_w  = (const float*)d_in[12];
    const float* mh_b  = (const float*)d_in[13];
    float* out = (float*)d_out;

    char* ws = (char*)d_ws;
    float* stepvec = (float*)(ws + OFF_STEPVEC);
    float* coefs   = (float*)(ws + OFF_COEFS);
    h16*   w1xT    = (h16*)(ws + OFF_W1XT);
    h16*   w2T     = (h16*)(ws + OFF_W2T);
    h16*   w3T     = (h16*)(ws + OFF_W3T);
    float* condc   = (float*)(ws + OFF_CONDC);
    float* mean    = (float*)(ws + OFF_MEAN);

    pre_time<<<1, 256, 0, stream>>>(te_w1, te_b1, te_w2, te_b2, dn_w1, dn_b1, stepvec, coefs);
    pre_weights<<<512, 256, 0, stream>>>(dn_w1, dn_w2, dn_w3, w1xT, w2T, w3T);
    pre_cond<<<768, 256, 0, stream>>>(x, dn_w1, mh_w, mh_b, condc, mean);

    const int lds_bytes = (256 * 128 + 256 * 256 + 128 * 256 + 64 * 128 + 64 * 256) * (int)sizeof(h16);
    static_assert((256 * 128 + 256 * 256 + 128 * 256 + 64 * 128 + 64 * 256) * 2 <= 320 * 1024, "LDS");
    (void)hipFuncSetAttribute((const void*)diff_main,
                              hipFuncAttributeMaxDynamicSharedMemorySize, lds_bytes);
    diff_main<<<MTOTAL / 64, 256, lds_bytes, stream>>>(
        stepvec, coefs, w1xT, w2T, w3T, condc, mean, dn_b2, dn_b3, out);
}
